// IntegratedCancerAnalysisModel_21921513079075
// MI455X (gfx1250) — compile-verified, matrix operand order checked by reading, not run on hardware
//
#include <hip/hip_runtime.h>
#include <hip/hip_bf16.h>

// ---------------------------------------------------------------------------
// MI455X (gfx1250, wave32) implementation of the multi-modal fusion model.
// All large GEMMs run on v_wmma_f32_16x16x32_bf16 (bf16 A/B, f32 accum).
// Weights are converted once per launch to bf16, transposed to [N][Kpad] so a
// wave's B fragment is two contiguous b128 loads per 16-col group.
// ---------------------------------------------------------------------------

typedef __bf16 bf16;
typedef __attribute__((ext_vector_type(16))) __bf16 v16bf;
typedef __attribute__((ext_vector_type(8)))  __bf16 v8bf;
typedef __attribute__((ext_vector_type(8)))  float   v8f;

static constexpr int  BN  = 16384;                 // batch
static constexpr int  EMB = 256;
static constexpr float BN_RSQ = 0.9999950000374997f; // 1/sqrt(1+1e-5)  (BatchNorm eval)
static constexpr float LNEPS  = 1e-5f;

enum { ACT_NONE = 0, ACT_RELU = 1, ACT_GELU = 2, ACT_TANH = 3 };

struct Epi {                       // GEMM epilogue descriptor
  const float* bias;               // [N]
  const float* bng;                // BatchNorm gamma (nullable)
  const float* bnb;                // BatchNorm beta
  float* outF; long long sF, oF;   // f32 output (nullable), row stride, col offset
  float* out2; long long s2, o2;   // second f32 output (nullable)
  bf16*  outB; long long sB, oB;   // bf16 output (nullable)
};

template<int ACT>
__device__ __forceinline__ float activate(float v) {
  if (ACT == ACT_RELU) return fmaxf(v, 0.0f);
  if (ACT == ACT_GELU) return 0.5f * v * (1.0f + erff(v * 0.70710678118654752f));
  if (ACT == ACT_TANH) return tanhf(v);
  return v;
}

// C fragment layout (16x16 f32, wave32): VGPR i, lanes 0-15 -> row i, lanes
// 16-31 -> row i+8; col = lane & 15.
template<int ACT>
__device__ __forceinline__ void epilogue(const v8f acc[4], long long row0, int n0,
                                         int r, int half, const Epi& ep) {
#pragma unroll
  for (int t = 0; t < 4; ++t) {
    const int col  = n0 + t * 16 + r;
    const float bs = ep.bias[col];
    const float gs = ep.bng ? ep.bng[col] * BN_RSQ : 1.0f;
    const float bb = ep.bnb ? ep.bnb[col] : 0.0f;
#pragma unroll
    for (int i = 0; i < 8; ++i) {
      const long long row = row0 + half * 8 + i;
      float v = activate<ACT>(acc[t][i] + bs);
      if (ep.bng) v = v * gs + bb;
      if (ep.outF) ep.outF[row * ep.sF + ep.oF + col] = v;
      if (ep.out2) ep.out2[row * ep.s2 + ep.o2 + col] = v;
      if (ep.outB) ep.outB[row * ep.sB + ep.oB + col] = (bf16)v;
    }
  }
}

// ---------------------------------------------------------------------------
// Weight prep: f32 [K][N] row-major -> bf16 [N][Kpad] (zero pad K -> Kpad).
// ---------------------------------------------------------------------------
__global__ void prep_wt(const float* __restrict__ W, bf16* __restrict__ WT,
                        int K, int N, int Kpad) {
  long long idx = (long long)blockIdx.x * 256 + threadIdx.x;
  if (idx >= (long long)N * Kpad) return;
  int n = (int)(idx / Kpad), k = (int)(idx % Kpad);
  WT[idx] = (k < K) ? (bf16)W[(long long)k * N + n] : (bf16)0.0f;
}

// ---------------------------------------------------------------------------
// GEMM, A in f32 (raw model inputs): staged through LDS with f32->bf16
// conversion and K zero-padding. Block = 256 thr = 8 waves -> 128x64 tile,
// each wave owns 16 rows x 64 cols (4 WMMA accumulators).
// ---------------------------------------------------------------------------
template<int ACT>
__global__ __launch_bounds__(256) void gemm_f32a(const float* __restrict__ A, int lda, int K,
                                                 const bf16* __restrict__ WT, int Kpad,
                                                 int N, Epi ep) {
  __shared__ __align__(16) bf16 As[128 * 40];     // 32 k-cols + 8 pad (80B stride)
  const int tid  = threadIdx.x;
  const int wave = tid >> 5, lane = tid & 31;
  const int half = lane >> 4, r = lane & 15;
  const int m0   = wave * 16;
  const long long rowBase = (long long)blockIdx.x * 128;
  const int n0   = blockIdx.y * 64;

  const int srow = tid >> 1;                      // staging: 2 threads per row
  const int skk  = (tid & 1) * 16;
  const float* abase = A + (rowBase + srow) * (long long)lda;

  v8f acc[4] = {};
  for (int k0 = 0; k0 < Kpad; k0 += 32) {
    // stage 128x32 f32 -> bf16 LDS tile
    union { bf16 e[16]; v8bf v[2]; } tmp;
    if (k0 + 32 <= K) {
      // uniform (scalar-branch) fast path: rows are 16B aligned for every lda
      // used here (512/2048/156/1000 are all multiples of 4) -> float4 loads.
      const float4* src = (const float4*)(abase + k0 + skk);
#pragma unroll
      for (int c = 0; c < 4; ++c) {
        float4 f = src[c];
        tmp.e[c*4+0] = (bf16)f.x; tmp.e[c*4+1] = (bf16)f.y;
        tmp.e[c*4+2] = (bf16)f.z; tmp.e[c*4+3] = (bf16)f.w;
      }
    } else {
      // K tail (only last k-step of cli K=156 / met K=1000): guarded scalar.
#pragma unroll
      for (int u = 0; u < 16; ++u) {
        int kg = k0 + skk + u;
        tmp.e[u] = (kg < K) ? (bf16)abase[kg] : (bf16)0.0f;
      }
    }
    {
      v8bf* dst = (v8bf*)&As[srow * 40 + skk];
      dst[0] = tmp.v[0]; dst[1] = tmp.v[1];
    }
    __syncthreads();
    // A fragment: lane<16 -> row m0+r, K {0..7,16..23}; lane>=16 -> K {8..15,24..31}
    union { v16bf v; v8bf h[2]; } af;
    const bf16* ap = &As[(m0 + r) * 40 + half * 8];
    af.h[0] = *(const v8bf*)ap;
    af.h[1] = *(const v8bf*)(ap + 16);
#pragma unroll
    for (int t = 0; t < 4; ++t) {
      union { v16bf v; v8bf h[2]; } bfr;     // B fragment from [N][Kpad] bf16
      const bf16* bp = WT + (long long)(n0 + t * 16 + r) * Kpad + k0 + half * 8;
      bfr.h[0] = *(const v8bf*)bp;
      bfr.h[1] = *(const v8bf*)(bp + 16);
      acc[t] = __builtin_amdgcn_wmma_f32_16x16x32_bf16(false, af.v, false, bfr.v,
                                                       (short)0, acc[t], false, false);
    }
    __syncthreads();
  }
  epilogue<ACT>(acc, rowBase + m0, n0, r, half, ep);
}

// ---------------------------------------------------------------------------
// GEMM, A already bf16 in global ([rows][lda], K multiple of 32): direct
// per-lane b128 fragment loads, no LDS / no barriers.
// ---------------------------------------------------------------------------
template<int ACT>
__global__ __launch_bounds__(256) void gemm_bf16a(const bf16* __restrict__ A, int lda,
                                                  const bf16* __restrict__ WT,
                                                  int K, int N, Epi ep) {
  const int tid  = threadIdx.x;
  const int wave = tid >> 5, lane = tid & 31;
  const int half = lane >> 4, r = lane & 15;
  const long long rowBase = (long long)blockIdx.x * 128;
  const int m0 = wave * 16, n0 = blockIdx.y * 64;
  const bf16* ap = A + (rowBase + m0 + r) * (long long)lda;

  v8f acc[4] = {};
  for (int k0 = 0; k0 < K; k0 += 32) {
    union { v16bf v; v8bf h[2]; } af;
    af.h[0] = *(const v8bf*)(ap + k0 + half * 8);
    af.h[1] = *(const v8bf*)(ap + k0 + half * 8 + 16);
#pragma unroll
    for (int t = 0; t < 4; ++t) {
      union { v16bf v; v8bf h[2]; } bfr;
      const bf16* bp = WT + (long long)(n0 + t * 16 + r) * K + k0 + half * 8;
      __builtin_prefetch((const void*)(bp + 32), 0, 1);   // next k-step of W
      bfr.h[0] = *(const v8bf*)bp;
      bfr.h[1] = *(const v8bf*)(bp + 16);
      acc[t] = __builtin_amdgcn_wmma_f32_16x16x32_bf16(false, af.v, false, bfr.v,
                                                       (short)0, acc[t], false, false);
    }
  }
  epilogue<ACT>(acc, rowBase + m0, n0, r, half, ep);
}

// ---------------------------------------------------------------------------
// Row-wise LayerNorm over 256 (optional residual), wave per row (wave32).
// ---------------------------------------------------------------------------
__global__ __launch_bounds__(256) void ln_kernel(const float* __restrict__ X,
                                                 const float* __restrict__ R,
                                                 const float* __restrict__ g,
                                                 const float* __restrict__ be,
                                                 float* __restrict__ outF,
                                                 bf16* __restrict__ outB) {
  const int wave = threadIdx.x >> 5, lane = threadIdx.x & 31;
  const long long row = (long long)blockIdx.x * 8 + wave;
  const float* xr = X + row * EMB;
  const float* rr = R ? R + row * EMB : nullptr;
  float x[8], s = 0.f, sq = 0.f;
#pragma unroll
  for (int t = 0; t < 8; ++t) {
    int c = t * 32 + lane;
    float v = xr[c] + (rr ? rr[c] : 0.f);
    x[t] = v; s += v; sq += v * v;
  }
#pragma unroll
  for (int o = 16; o >= 1; o >>= 1) { s += __shfl_xor(s, o, 32); sq += __shfl_xor(sq, o, 32); }
  const float mu  = s * (1.0f / EMB);
  const float var = sq * (1.0f / EMB) - mu * mu;
  const float inv = rsqrtf(var + LNEPS);
#pragma unroll
  for (int t = 0; t < 8; ++t) {
    int c = t * 32 + lane;
    float y = g[c] * (x[t] - mu) * inv + be[c];
    if (outF) outF[row * EMB + c] = y;
    if (outB) outB[row * EMB + c] = (bf16)y;
  }
}

// ---------------------------------------------------------------------------
// Fused cross-modal attention: one wave per batch element b. Lane = (i,h)
// (query modality x head). Computes all 32x32 scores, blockwise (per source
// modality j) softmax, and the mean-over-j attention output — without ever
// materializing the [M,M,B,H,H] tensor (saves ~128 MB of HBM traffic).
// ---------------------------------------------------------------------------
__global__ __launch_bounds__(128) void attn_fused_kernel(const bf16* __restrict__ Qb,
                                                         const bf16* __restrict__ Kb,
                                                         const bf16* __restrict__ Vb,
                                                         bf16* __restrict__ attB) {
  __shared__ __align__(16) float Ks[4][32][36];
  __shared__ __align__(16) float Vs[4][32][36];
  const int wave = threadIdx.x >> 5, lane = threadIdx.x & 31;
  const long long b = (long long)blockIdx.x * 4 + wave;
  const int i = lane >> 3, h = lane & 7;          // also (j,g) for staging K/V
  const long long base = ((long long)i * BN + b) * EMB + h * 32;

  {   // stage this b's K and V (32 rows of 32) into LDS, and Q into registers
    union { v8bf v; bf16 e[8]; } t;
    const bf16* kp = Kb + base;
    const bf16* vp = Vb + base;
#pragma unroll
    for (int c = 0; c < 4; ++c) {
      t.v = *(const v8bf*)(kp + c * 8);
#pragma unroll
      for (int u = 0; u < 8; ++u) Ks[wave][lane][c * 8 + u] = (float)t.e[u];
      t.v = *(const v8bf*)(vp + c * 8);
#pragma unroll
      for (int u = 0; u < 8; ++u) Vs[wave][lane][c * 8 + u] = (float)t.e[u];
    }
  }
  float q[32];
  {
    union { v8bf v; bf16 e[8]; } t;
    const bf16* qp = Qb + base;
#pragma unroll
    for (int c = 0; c < 4; ++c) {
      t.v = *(const v8bf*)(qp + c * 8);
#pragma unroll
      for (int u = 0; u < 8; ++u) q[c * 8 + u] = (float)t.e[u];
    }
  }
  __syncthreads();

  const float inv_scale = 0.17677669529663689f;   // 1/sqrt(32)
  float a[32];
#pragma unroll
  for (int jg = 0; jg < 32; ++jg) {
    const float4* kr = (const float4*)&Ks[wave][jg][0];
    float acc = 0.f;
#pragma unroll
    for (int c = 0; c < 8; ++c) {
      float4 kv = kr[c];
      acc += q[c*4+0]*kv.x + q[c*4+1]*kv.y + q[c*4+2]*kv.z + q[c*4+3]*kv.w;
    }
    a[jg] = acc * inv_scale;
  }
#pragma unroll
  for (int j = 0; j < 4; ++j) {                   // softmax over g within each j
    float mx = a[j*8];
#pragma unroll
    for (int g = 1; g < 8; ++g) mx = fmaxf(mx, a[j*8+g]);
    float sum = 0.f;
#pragma unroll
    for (int g = 0; g < 8; ++g) { float e = __expf(a[j*8+g] - mx); a[j*8+g] = e; sum += e; }
    float rn = 0.25f / sum;                       // fold the mean over j (M=4)
#pragma unroll
    for (int g = 0; g < 8; ++g) a[j*8+g] *= rn;
  }
  float att[32];
#pragma unroll
  for (int d = 0; d < 32; ++d) att[d] = 0.f;
#pragma unroll
  for (int jg = 0; jg < 32; ++jg) {
    const float4* vr = (const float4*)&Vs[wave][jg][0];
    const float w = a[jg];
#pragma unroll
    for (int c = 0; c < 8; ++c) {
      float4 vv = vr[c];
      att[c*4+0] += w*vv.x; att[c*4+1] += w*vv.y; att[c*4+2] += w*vv.z; att[c*4+3] += w*vv.w;
    }
  }
  union { v8bf v; bf16 e[8]; } o;
  bf16* op = attB + base;
#pragma unroll
  for (int c = 0; c < 4; ++c) {
#pragma unroll
    for (int u = 0; u < 8; ++u) o.e[u] = (bf16)att[c*8+u];
    *(v8bf*)(op + c * 8) = o.v;
  }
}

// mean over the 4 modality slices -> attention_fused
__global__ void mean4_kernel(const float* __restrict__ A, float* __restrict__ out) {
  long long idx = (long long)blockIdx.x * 256 + threadIdx.x;
  const long long BE = (long long)BN * EMB;
  out[idx] = 0.25f * (A[idx] + A[BE + idx] + A[2*BE + idx] + A[3*BE + idx]);
}

// C = softmax(A[B,256](bf16) @ W[256,NS](f32) + b), wave per row.
template<int NS>
__global__ __launch_bounds__(256) void small_softmax_kernel(const bf16* __restrict__ A,
                                                            const float* __restrict__ W,
                                                            const float* __restrict__ bias,
                                                            float* __restrict__ out) {
  const int wave = threadIdx.x >> 5, lane = threadIdx.x & 31;
  const long long row = (long long)blockIdx.x * 8 + wave;
  union { v8bf v; bf16 e[8]; } t;
  t.v = *(const v8bf*)(A + row * EMB + lane * 8);
  float acc[NS];
#pragma unroll
  for (int n = 0; n < NS; ++n) acc[n] = 0.f;
#pragma unroll
  for (int kk = 0; kk < 8; ++kk) {
    const float av = (float)t.e[kk];
    const float* wr = W + (lane * 8 + kk) * NS;
#pragma unroll
    for (int n = 0; n < NS; ++n) acc[n] += av * wr[n];
  }
#pragma unroll
  for (int o = 16; o >= 1; o >>= 1)
#pragma unroll
    for (int n = 0; n < NS; ++n) acc[n] += __shfl_xor(acc[n], o, 32);
  float mx = -3.4e38f;
#pragma unroll
  for (int n = 0; n < NS; ++n) { acc[n] += bias[n]; mx = fmaxf(mx, acc[n]); }
  float sum = 0.f;
#pragma unroll
  for (int n = 0; n < NS; ++n) { acc[n] = __expf(acc[n] - mx); sum += acc[n]; }
  if (lane < NS) out[row * NS + lane] = acc[lane] / sum;
}

// fused = s0*attn_fused + s1*sum_m gate[m]*T2[m] + s2*concat_proj
__global__ void combine_kernel(const float* __restrict__ strat, const float* __restrict__ gate,
                               const float* __restrict__ attF, const float* __restrict__ T2,
                               const float* __restrict__ projout, float* __restrict__ out) {
  long long idx = (long long)blockIdx.x * 256 + threadIdx.x;
  long long b = idx >> 8;
  const long long BE = (long long)BN * EMB;
  float gsum = gate[b*4+0]*T2[idx] + gate[b*4+1]*T2[BE+idx]
             + gate[b*4+2]*T2[2*BE+idx] + gate[b*4+3]*T2[3*BE+idx];
  out[idx] = strat[b*3+0]*attF[idx] + strat[b*3+1]*gsum + strat[b*3+2]*projout[idx];
}

// ---------------------------------------------------------------------------
// Host-side dispatch helpers
// ---------------------------------------------------------------------------
static void run_gemm_f32a(hipStream_t s, int act, const float* A, int lda, int K,
                          const bf16* WT, int Kpad, int N, const Epi& ep) {
  dim3 grid(BN / 128, N / 64);
  switch (act) {
    case ACT_NONE: gemm_f32a<ACT_NONE><<<grid, 256, 0, s>>>(A, lda, K, WT, Kpad, N, ep); break;
    case ACT_RELU: gemm_f32a<ACT_RELU><<<grid, 256, 0, s>>>(A, lda, K, WT, Kpad, N, ep); break;
    case ACT_GELU: gemm_f32a<ACT_GELU><<<grid, 256, 0, s>>>(A, lda, K, WT, Kpad, N, ep); break;
    default:       gemm_f32a<ACT_TANH><<<grid, 256, 0, s>>>(A, lda, K, WT, Kpad, N, ep); break;
  }
}
static void run_gemm_bf16a(hipStream_t s, int act, const bf16* A, int lda,
                           const bf16* WT, int K, int N, const Epi& ep) {
  dim3 grid(BN / 128, N / 64);
  switch (act) {
    case ACT_NONE: gemm_bf16a<ACT_NONE><<<grid, 256, 0, s>>>(A, lda, WT, K, N, ep); break;
    case ACT_RELU: gemm_bf16a<ACT_RELU><<<grid, 256, 0, s>>>(A, lda, WT, K, N, ep); break;
    case ACT_GELU: gemm_bf16a<ACT_GELU><<<grid, 256, 0, s>>>(A, lda, WT, K, N, ep); break;
    default:       gemm_bf16a<ACT_TANH><<<grid, 256, 0, s>>>(A, lda, WT, K, N, ep); break;
  }
}

// d_in flat order: inputs (setup_inputs dict order) then params in dict
// insertion order (wsi, gen, cli, met, attn, gate, strat, proj).
enum {
  IN_XWSI = 0, IN_XGEN, IN_XCLI, IN_XMET,
  P_WSI_W1, P_WSI_B1, P_WSI_G1, P_WSI_BE1, P_WSI_W2, P_WSI_B2, P_WSI_G2, P_WSI_BE2, P_WSI_WO, P_WSI_BO,
  P_GEN_W1, P_GEN_B1, P_GEN_G1, P_GEN_BE1, P_GEN_W2, P_GEN_B2, P_GEN_G2, P_GEN_BE2, P_GEN_WO, P_GEN_BO,
  P_CLI_W1, P_CLI_B1, P_CLI_G1, P_CLI_BE1, P_CLI_W2, P_CLI_B2, P_CLI_WO, P_CLI_BO,
  P_MET_W1, P_MET_B1, P_MET_G1, P_MET_BE1, P_MET_W2, P_MET_B2, P_MET_WO, P_MET_BO,
  P_AT_WQ, P_AT_BQ, P_AT_WK, P_AT_BK, P_AT_WV, P_AT_BV, P_AT_WO, P_AT_BO, P_AT_G, P_AT_BE,
  P_GA_WG1, P_GA_BG1, P_GA_WG2, P_GA_BG2, P_GA_WT1, P_GA_BT1, P_GA_WT2, P_GA_BT2,
  P_ST_W1, P_ST_B1, P_ST_W2, P_ST_B2,
  P_PJ_W1, P_PJ_B1, P_PJ_W2, P_PJ_B2
};

extern "C" void kernel_launch(void* const* d_in, const int* in_sizes, int n_in,
                              void* d_out, int out_size, void* d_ws, size_t ws_size,
                              hipStream_t stream) {
  (void)in_sizes; (void)n_in; (void)out_size; (void)ws_size;
  const long long BE = (long long)BN * EMB;
  char* wsb = (char*)d_ws; size_t cur = 0;
  auto alloc = [&](size_t bytes) -> void* {
    void* p = wsb + cur; cur += (bytes + 255) & ~(size_t)255; return p;
  };
  auto in = [&](int i) { return (const float*)d_in[i]; };
  auto prep = [&](const float* src, int K, int N, int Kpad) -> bf16* {
    bf16* dst = (bf16*)alloc((size_t)N * Kpad * sizeof(bf16));
    long long tot = (long long)N * Kpad;
    prep_wt<<<(unsigned)((tot + 255) / 256), 256, 0, stream>>>(src, dst, K, N, Kpad);
    return dst;
  };

  // ---- weight conversion (f32 [K][N] -> bf16 [N][Kpad]) ----
  bf16* wsiW1T = prep(in(P_WSI_W1), 512, 512, 512);
  bf16* wsiW2T = prep(in(P_WSI_W2), 512, 256, 512);
  bf16* wsiWoT = prep(in(P_WSI_WO), 256, 256, 256);
  bf16* genW1T = prep(in(P_GEN_W1), 2048, 256, 2048);
  bf16* genW2T = prep(in(P_GEN_W2), 256, 256, 256);
  bf16* genWoT = prep(in(P_GEN_WO), 256, 256, 256);
  bf16* cliW1T = prep(in(P_CLI_W1), 156, 256, 160);
  bf16* cliW2T = prep(in(P_CLI_W2), 256, 256, 256);
  bf16* cliWoT = prep(in(P_CLI_WO), 256, 256, 256);
  bf16* metW1T = prep(in(P_MET_W1), 1000, 256, 1024);
  bf16* metW2T = prep(in(P_MET_W2), 256, 256, 256);
  bf16* metWoT = prep(in(P_MET_WO), 256, 256, 256);
  bf16 *WqT[4], *WkT[4], *WvT[4], *WaoT[4], *Wt1T[4], *Wt2T[4];
  for (int m = 0; m < 4; ++m) {
    WqT[m]  = prep(in(P_AT_WQ)  + (long long)m * 65536, 256, 256, 256);
    WkT[m]  = prep(in(P_AT_WK)  + (long long)m * 65536, 256, 256, 256);
    WvT[m]  = prep(in(P_AT_WV)  + (long long)m * 65536, 256, 256, 256);
    WaoT[m] = prep(in(P_AT_WO)  + (long long)m * 65536, 256, 256, 256);
    Wt1T[m] = prep(in(P_GA_WT1) + (long long)m * 65536, 256, 256, 256);
    Wt2T[m] = prep(in(P_GA_WT2) + (long long)m * 65536, 256, 256, 256);
  }
  bf16* Wg1T = prep(in(P_GA_WG1), 1024, 256, 1024);
  bf16* stW1T = prep(in(P_ST_W1), 1024, 256, 1024);
  bf16* pjW1T = prep(in(P_PJ_W1), 1024, 256, 1024);
  bf16* pjW2T = prep(in(P_PJ_W2), 256, 256, 256);

  // ---- activation workspace ----
  bf16*  hbA     = (bf16*) alloc((size_t)BN * 512 * 2);
  bf16*  hbB     = (bf16*) alloc((size_t)BN * 256 * 2);
  float* hf      = (float*)alloc((size_t)BN * 256 * 4);
  bf16*  concatB = (bf16*) alloc((size_t)BN * 1024 * 2);
  float* F32     = (float*)alloc((size_t)4 * BE * 4);
  bf16*  Qb      = (bf16*) alloc((size_t)4 * BE * 2);
  bf16*  Kb      = (bf16*) alloc((size_t)4 * BE * 2);
  bf16*  Vb      = (bf16*) alloc((size_t)4 * BE * 2);
  bf16*  attB    = (bf16*) alloc((size_t)4 * BE * 2);
  float* attnout = (float*)alloc((size_t)4 * BE * 4);
  float* attnF   = (float*)alloc((size_t)BE * 4);
  bf16*  g1b     = (bf16*) alloc((size_t)BN * 256 * 2);
  bf16*  s1b     = (bf16*) alloc((size_t)BN * 256 * 2);
  bf16*  pjh     = (bf16*) alloc((size_t)BN * 256 * 2);
  float* projout = (float*)alloc((size_t)BE * 4);
  float* gateP   = (float*)alloc((size_t)BN * 4 * 4);
  float* stratP  = (float*)alloc((size_t)BN * 3 * 4);
  bf16*  Tb = Qb;     // Q is dead after the attention kernel
  float* T2 = F32;    // F is dead after the residual LayerNorms

  // ---- wsi encoder: relu+BN, relu+BN, linear ----
  { Epi e{}; e.bias=in(P_WSI_B1); e.bng=in(P_WSI_G1); e.bnb=in(P_WSI_BE1);
    e.outB=hbA; e.sB=512;
    run_gemm_f32a(stream, ACT_RELU, in(IN_XWSI), 512, 512, wsiW1T, 512, 512, e); }
  { Epi e{}; e.bias=in(P_WSI_B2); e.bng=in(P_WSI_G2); e.bnb=in(P_WSI_BE2);
    e.outB=hbB; e.sB=256;
    run_gemm_bf16a(stream, ACT_RELU, hbA, 512, wsiW2T, 512, 256, e); }
  { Epi e{}; e.bias=in(P_WSI_BO);
    e.outF=F32; e.sF=256; e.outB=concatB; e.sB=1024; e.oB=0;
    run_gemm_bf16a(stream, ACT_NONE, hbB, 256, wsiWoT, 256, 256, e); }

  // ---- gen encoder: gelu+LN, gelu+LN, linear ----
  { Epi e{}; e.bias=in(P_GEN_B1); e.outF=hf; e.sF=256;
    run_gemm_f32a(stream, ACT_GELU, in(IN_XGEN), 2048, 2048, genW1T, 2048, 256, e); }
  ln_kernel<<<BN/8, 256, 0, stream>>>(hf, nullptr, in(P_GEN_G1), in(P_GEN_BE1), nullptr, hbA);
  { Epi e{}; e.bias=in(P_GEN_B2); e.outF=hf; e.sF=256;
    run_gemm_bf16a(stream, ACT_GELU, hbA, 256, genW2T, 256, 256, e); }
  ln_kernel<<<BN/8, 256, 0, stream>>>(hf, nullptr, in(P_GEN_G2), in(P_GEN_BE2), nullptr, hbB);
  { Epi e{}; e.bias=in(P_GEN_BO);
    e.outF=F32 + BE; e.sF=256; e.outB=concatB; e.sB=1024; e.oB=256;
    run_gemm_bf16a(stream, ACT_NONE, hbB, 256, genWoT, 256, 256, e); }

  // ---- cli encoder: relu+BN, relu, linear ----
  { Epi e{}; e.bias=in(P_CLI_B1); e.bng=in(P_CLI_G1); e.bnb=in(P_CLI_BE1);
    e.outB=hbA; e.sB=256;
    run_gemm_f32a(stream, ACT_RELU, in(IN_XCLI), 156, 156, cliW1T, 160, 256, e); }
  { Epi e{}; e.bias=in(P_CLI_B2); e.outB=hbB; e.sB=256;
    run_gemm_bf16a(stream, ACT_RELU, hbA, 256, cliW2T, 256, 256, e); }
  { Epi e{}; e.bias=in(P_CLI_BO);
    e.outF=F32 + 2*BE; e.sF=256; e.outB=concatB; e.sB=1024; e.oB=512;
    run_gemm_bf16a(stream, ACT_NONE, hbB, 256, cliWoT, 256, 256, e); }

  // ---- met encoder: tanh+LN, tanh, linear ----
  { Epi e{}; e.bias=in(P_MET_B1); e.outF=hf; e.sF=256;
    run_gemm_f32a(stream, ACT_TANH, in(IN_XMET), 1000, 1000, metW1T, 1024, 256, e); }
  ln_kernel<<<BN/8, 256, 0, stream>>>(hf, nullptr, in(P_MET_G1), in(P_MET_BE1), nullptr, hbA);
  { Epi e{}; e.bias=in(P_MET_B2); e.outB=hbB; e.sB=256;
    run_gemm_bf16a(stream, ACT_TANH, hbA, 256, metW2T, 256, 256, e); }
  { Epi e{}; e.bias=in(P_MET_BO);
    e.outF=F32 + 3*BE; e.sF=256; e.outB=concatB; e.sB=1024; e.oB=768;
    run_gemm_bf16a(stream, ACT_NONE, hbB, 256, metWoT, 256, 256, e); }

  // ---- attention: QKV projections (A = feature columns of concat) ----
  for (int m = 0; m < 4; ++m) {
    { Epi e{}; e.bias=in(P_AT_BQ)+m*256; e.outB=Qb+(size_t)m*BE; e.sB=256;
      run_gemm_bf16a(stream, ACT_NONE, concatB + m*256, 1024, WqT[m], 256, 256, e); }
    { Epi e{}; e.bias=in(P_AT_BK)+m*256; e.outB=Kb+(size_t)m*BE; e.sB=256;
      run_gemm_bf16a(stream, ACT_NONE, concatB + m*256, 1024, WkT[m], 256, 256, e); }
    { Epi e{}; e.bias=in(P_AT_BV)+m*256; e.outB=Vb+(size_t)m*BE; e.sB=256;
      run_gemm_bf16a(stream, ACT_NONE, concatB + m*256, 1024, WvT[m], 256, 256, e); }
  }
  attn_fused_kernel<<<BN/4, 128, 0, stream>>>(Qb, Kb, Vb, attB);
  for (int m = 0; m < 4; ++m) {   // output projection + residual LayerNorm
    { Epi e{}; e.bias=in(P_AT_BO)+m*256; e.outF=attnout+(size_t)m*BE; e.sF=256;
      run_gemm_bf16a(stream, ACT_NONE, attB + (size_t)m*BE, 256, WaoT[m], 256, 256, e); }
    ln_kernel<<<BN/8, 256, 0, stream>>>(attnout + (size_t)m*BE, F32 + (size_t)m*BE,
                                        in(P_AT_G)+m*256, in(P_AT_BE)+m*256,
                                        attnout + (size_t)m*BE, nullptr);
  }
  mean4_kernel<<<(unsigned)(BE/256), 256, 0, stream>>>(attnout, attnF);

  // ---- gated fusion + strategy selector + concat projection ----
  { Epi e{}; e.bias=in(P_GA_BG1); e.outB=g1b; e.sB=256;
    run_gemm_bf16a(stream, ACT_RELU, concatB, 1024, Wg1T, 1024, 256, e); }
  small_softmax_kernel<4><<<BN/8, 256, 0, stream>>>(g1b, in(P_GA_WG2), in(P_GA_BG2), gateP);
  { Epi e{}; e.bias=in(P_ST_B1); e.outB=s1b; e.sB=256;
    run_gemm_bf16a(stream, ACT_RELU, concatB, 1024, stW1T, 1024, 256, e); }
  small_softmax_kernel<3><<<BN/8, 256, 0, stream>>>(s1b, in(P_ST_W2), in(P_ST_B2), stratP);
  { Epi e{}; e.bias=in(P_PJ_B1); e.outB=pjh; e.sB=256;
    run_gemm_bf16a(stream, ACT_RELU, concatB, 1024, pjW1T, 1024, 256, e); }
  { Epi e{}; e.bias=in(P_PJ_B2); e.outF=projout; e.sF=256;
    run_gemm_bf16a(stream, ACT_NONE, pjh, 256, pjW2T, 256, 256, e); }
  for (int m = 0; m < 4; ++m) {   // T = relu(F@Wt1+bt1)@Wt2+bt2
    { Epi e{}; e.bias=in(P_GA_BT1)+m*256; e.outB=Tb+(size_t)m*BE; e.sB=256;
      run_gemm_bf16a(stream, ACT_RELU, concatB + m*256, 1024, Wt1T[m], 256, 256, e); }
    { Epi e{}; e.bias=in(P_GA_BT2)+m*256; e.outF=T2+(size_t)m*BE; e.sF=256;
      run_gemm_bf16a(stream, ACT_NONE, Tb + (size_t)m*BE, 256, Wt2T[m], 256, 256, e); }
  }

  combine_kernel<<<(unsigned)(BE/256), 256, 0, stream>>>(stratP, gateP, attnF, T2,
                                                         projout, (float*)d_out);
}
